// LSTMMirrored_2637109920175
// MI455X (gfx1250) — matrix operand order repaired and checked by reading, not compiled, not run on hardware
//
#include <hip/hip_runtime.h>

// ---------------------------------------------------------------------------
// Fused LSTM-gate kernel for MI455X (gfx1250, wave32, WMMA bf16).
//   pre_g = X @ Wgx + H @ Wgh + b_g   for g in {f,i,o,c}
//   f,i,o = sigmoid(pre), g = tanh(pre_c)
//   c = f*carousel + i*g ; h = o*c
//
// Fast path (needs ~80MB workspace):
//   pass 1: X,H f32 -> bf16 (elementwise)
//   pass 2: 8 weight matrices f32 [k][n] -> bf16 [n][k] (tiled transpose)
//   pass 3: double-buffered WMMA GEMM with GLOBAL_LOAD_ASYNC_TO_LDS_B128
//           staging (ASYNCcnt-pipelined) + fused activation epilogue
// Fallback path (no workspace): converts f32->bf16 during LDS staging.
// ---------------------------------------------------------------------------

typedef __bf16 bf16_t;
typedef bf16_t v16bf __attribute__((ext_vector_type(16)));
typedef bf16_t v8bf  __attribute__((ext_vector_type(8)));
typedef float  v8f   __attribute__((ext_vector_type(8)));

#define B_ROWS 16384
#define D_IN   1024
#define D_OUT  1024

#define BM 128          // block tile M
#define BN 64           // block tile N (per gate)
#define BK 32           // K step (one wmma_..x32 chain step)
#define LDK 40          // padded LDS row length (80B rows -> spreads banks)

#define WMMA_BF16(a, b, c) \
  __builtin_amdgcn_wmma_f32_16x16x32_bf16(false, (a), false, (b), (short)0, (c), false, false)

__device__ __forceinline__ v16bf cat8(v8bf lo, v8bf hi) {
  return __builtin_shufflevector(lo, hi, 0, 1, 2, 3, 4, 5, 6, 7,
                                         8, 9, 10, 11, 12, 13, 14, 15);
}

__device__ __forceinline__ float sigmoid_f(float x) {
  return 1.0f / (1.0f + __expf(-x));
}

// --- CDNA5 async staging primitives (ISA 08_async_tensor: GVS addressing) ---
__device__ __forceinline__ void async_b128_to_lds(unsigned lds_addr,
                                                  const void* sbase,
                                                  unsigned voffset_bytes) {
  // VDST = LDS byte address, VADDR = 32-bit byte offset, SADDR = 64-bit base
  asm volatile("global_load_async_to_lds_b128 %0, %1, %2 offset:0"
               :: "v"(lds_addr), "v"(voffset_bytes), "s"(sbase)
               : "memory");
}
__device__ __forceinline__ void wait_asynccnt_le12() {
  asm volatile("s_wait_asynccnt 0xc" ::: "memory");
}
__device__ __forceinline__ void wait_asynccnt_0() {
  asm volatile("s_wait_asynccnt 0x0" ::: "memory");
}
__device__ __forceinline__ unsigned lds_addr_u32(const void* p) {
  // generic LDS address: addr[31:0] is the LDS byte offset (ISA 10.2 aperture map)
  return (unsigned)(unsigned long long)p;
}

// ---------------------------------------------------------------------------
// Pass 1: elementwise f32 -> bf16 (8 elements per thread)
// ---------------------------------------------------------------------------
__global__ __launch_bounds__(256)
void cvt_f32_to_bf16(const float* __restrict__ src, bf16_t* __restrict__ dst, int n8) {
  const int i = blockIdx.x * blockDim.x + threadIdx.x;
  if (i >= n8) return;
  const float4* s = (const float4*)src + (size_t)i * 2;
  const float4 a = s[0];
  const float4 b = s[1];
  v8bf o;
  o[0] = (bf16_t)a.x; o[1] = (bf16_t)a.y; o[2] = (bf16_t)a.z; o[3] = (bf16_t)a.w;
  o[4] = (bf16_t)b.x; o[5] = (bf16_t)b.y; o[6] = (bf16_t)b.z; o[7] = (bf16_t)b.w;
  *(v8bf*)(dst + (size_t)i * 8) = o;
}

// ---------------------------------------------------------------------------
// Pass 2: weight transpose-convert: f32 [k][n] -> bf16 [n][k], 64x64 tiles
// grid = (D_OUT/64, D_IN/64, 8)
// ---------------------------------------------------------------------------
#define TP 80  // padded LDS row (160B, keeps 16B-aligned chunks)
__global__ __launch_bounds__(256)
void wt_transpose_bf16(const float* __restrict__ Wfx, const float* __restrict__ Wfh,
                       const float* __restrict__ Wix, const float* __restrict__ Wih,
                       const float* __restrict__ Wox, const float* __restrict__ Woh,
                       const float* __restrict__ Wcx, const float* __restrict__ Wch,
                       bf16_t* __restrict__ WtAll) {
  __shared__ bf16_t T[64][TP];
  const int tid = threadIdx.x;
  const int n0 = blockIdx.x * 64;
  const int k0 = blockIdx.y * 64;
  const int w  = blockIdx.z;

  const float* Wp;
  switch (w) {
    case 0: Wp = Wfx; break; case 1: Wp = Wfh; break;
    case 2: Wp = Wix; break; case 3: Wp = Wih; break;
    case 4: Wp = Wox; break; case 5: Wp = Woh; break;
    case 6: Wp = Wcx; break; default: Wp = Wch; break;
  }
  bf16_t* Dst = WtAll + (size_t)w * D_IN * D_OUT;

#pragma unroll
  for (int p = 0; p < 4; ++p) {
    const int idx = p * 256 + tid;      // 0..1023
    const int kk  = idx >> 4;           // 0..63
    const int nn  = (idx & 15) * 4;     // 0..60
    const float4 v = *(const float4*)(Wp + (size_t)(k0 + kk) * D_OUT + n0 + nn);
    T[nn + 0][kk] = (bf16_t)v.x;
    T[nn + 1][kk] = (bf16_t)v.y;
    T[nn + 2][kk] = (bf16_t)v.z;
    T[nn + 3][kk] = (bf16_t)v.w;
  }
  __syncthreads();

#pragma unroll
  for (int p = 0; p < 2; ++p) {
    const int idx = p * 256 + tid;      // 0..511
    const int n   = idx >> 3;           // 0..63
    const int c8  = (idx & 7) * 8;      // 0..56
    const v8bf v = *(const v8bf*)&T[n][c8];
    *(v8bf*)(Dst + (size_t)(n0 + n) * D_IN + k0 + c8) = v;
  }
}

// ---------------------------------------------------------------------------
// Shared compute step: one BK=32 slab of WMMAs over staged LDS tiles
// ---------------------------------------------------------------------------
__device__ __forceinline__ void wmma_compute_step(
    const bf16_t (* __restrict__ Xs)[LDK],
    const bf16_t (* __restrict__ Hs)[LDK],
    const bf16_t (* __restrict__ Ws)[BN][LDK],
    const int wave_m, const int wave_n, const int lane15,
    const int khalfA, const int kbB,
    v8f (&acc)[4][2][2]) {
  v16bf ax[2], ah[2];
#pragma unroll
  for (int mi = 0; mi < 2; ++mi) {
    const int r = wave_m * 32 + mi * 16 + lane15;
    v8bf lo = *(const v8bf*)&Xs[r][khalfA];
    v8bf hi = *(const v8bf*)&Xs[r][khalfA + 8];
    ax[mi] = cat8(lo, hi);
    lo = *(const v8bf*)&Hs[r][khalfA];
    hi = *(const v8bf*)&Hs[r][khalfA + 8];
    ah[mi] = cat8(lo, hi);
  }
#pragma unroll
  for (int g = 0; g < 4; ++g) {
    v16bf bx[2], bh[2];
#pragma unroll
    for (int ni = 0; ni < 2; ++ni) {
      const int c = wave_n * 32 + ni * 16 + lane15;
      v8bf lo = *(const v8bf*)&Ws[2 * g + 0][c][kbB];
      v8bf hi = *(const v8bf*)&Ws[2 * g + 0][c][kbB + 8];
      bx[ni] = cat8(lo, hi);
      lo = *(const v8bf*)&Ws[2 * g + 1][c][kbB];
      hi = *(const v8bf*)&Ws[2 * g + 1][c][kbB + 8];
      bh[ni] = cat8(lo, hi);
    }
#pragma unroll
    for (int mi = 0; mi < 2; ++mi) {
#pragma unroll
      for (int ni = 0; ni < 2; ++ni) {
        acc[g][mi][ni] = WMMA_BF16(ax[mi], bx[ni], acc[g][mi][ni]);
        acc[g][mi][ni] = WMMA_BF16(ah[mi], bh[ni], acc[g][mi][ni]);
      }
    }
  }
}

// ---------------------------------------------------------------------------
// Shared epilogue: bias + activations + elementwise, write h and c
// ---------------------------------------------------------------------------
__device__ __forceinline__ void lstm_epilogue(
    const v8f (&acc)[4][2][2],
    const float* __restrict__ bf_, const float* __restrict__ bi_,
    const float* __restrict__ bo_, const float* __restrict__ bc_,
    const int* __restrict__ carousel,
    float* __restrict__ outH, float* __restrict__ outC,
    const int m0, const int n0, const int wave_m, const int wave_n,
    const int lane15, const int hi_half) {
  const float carF = (float)carousel[0];
#pragma unroll
  for (int ni = 0; ni < 2; ++ni) {
    const int col = n0 + wave_n * 32 + ni * 16 + lane15;
    const float bfv = bf_[col];
    const float biv = bi_[col];
    const float bov = bo_[col];
    const float bcv = bc_[col];
#pragma unroll
    for (int mi = 0; mi < 2; ++mi) {
#pragma unroll
      for (int r = 0; r < 8; ++r) {
        const int row = m0 + wave_m * 32 + mi * 16 + r + (hi_half << 3);
        const float fv = sigmoid_f(acc[0][mi][ni][r] + bfv);
        const float iv = sigmoid_f(acc[1][mi][ni][r] + biv);
        const float ov = sigmoid_f(acc[2][mi][ni][r] + bov);
        const float gv = tanhf(acc[3][mi][ni][r] + bcv);
        const float cv = fv * carF + iv * gv;
        const float hv = ov * cv;
        const size_t off = (size_t)row * D_OUT + col;
        outH[off] = hv;
        outC[off] = cv;
      }
    }
  }
}

// ---------------------------------------------------------------------------
// Async tile staging: 12 x b128 async copies per thread (X:2, H:2, W:8)
// ---------------------------------------------------------------------------
__device__ __forceinline__ void stage_async_tile(
    bf16_t (* __restrict__ Xs)[LDK], bf16_t (* __restrict__ Hs)[LDK],
    bf16_t (* __restrict__ Ws)[BN][LDK],
    const bf16_t* __restrict__ Xbf, const bf16_t* __restrict__ Hbf,
    const bf16_t* __restrict__ WtAll,
    const int tid, const int m0, const int n0, const int k0) {
#pragma unroll
  for (int p = 0; p < 2; ++p) {
    const int idx = p * 256 + tid;    // 0..511
    const int row = idx >> 2;         // 0..127
    const int c8  = (idx & 3) * 8;    // 0,8,16,24
    async_b128_to_lds(lds_addr_u32(&Xs[row][c8]), Xbf,
                      (unsigned)(((m0 + row) * D_IN + k0 + c8) * 2));
    async_b128_to_lds(lds_addr_u32(&Hs[row][c8]), Hbf,
                      (unsigned)(((m0 + row) * D_OUT + k0 + c8) * 2));
  }
  const int row = tid >> 2;           // 0..63
  const int c8  = (tid & 3) * 8;      // 0,8,16,24
#pragma unroll
  for (int w = 0; w < 8; ++w) {
    async_b128_to_lds(lds_addr_u32(&Ws[w][row][c8]), WtAll,
                      (unsigned)((w * D_IN * D_OUT + (n0 + row) * D_IN + k0 + c8) * 2));
  }
}

// ---------------------------------------------------------------------------
// Pass 3 (fast path): double-buffered async-staged WMMA GEMM + epilogue
// ---------------------------------------------------------------------------
__global__ __launch_bounds__(256)
void lstm_fused_wmma_async(const int* __restrict__ carousel,
                           const float* __restrict__ bf_, const float* __restrict__ bi_,
                           const float* __restrict__ bo_, const float* __restrict__ bc_,
                           const bf16_t* __restrict__ Xbf,   // [B_ROWS][D_IN]
                           const bf16_t* __restrict__ Hbf,   // [B_ROWS][D_OUT]
                           const bf16_t* __restrict__ WtAll, // 8 x [D_OUT][D_IN]
                           float* __restrict__ outH, float* __restrict__ outC) {
  __shared__ bf16_t Xs[2][BM][LDK];
  __shared__ bf16_t Hs[2][BM][LDK];
  __shared__ bf16_t Ws[2][8][BN][LDK];

  const int tid  = threadIdx.x;
  const int wid  = tid >> 5;
  const int lane = tid & 31;
  const int lane15 = lane & 15;
  const int hi_half = lane >> 4;
  const int wave_m = wid & 3;
  const int wave_n = wid >> 2;
  const int m0 = blockIdx.y * BM;
  const int n0 = blockIdx.x * BN;

  const int khalfA = hi_half * 16;  // A-fragment K offset (contiguous k = 16h + s, matches B)
  const int kbB    = hi_half * 16;  // B-fragment K offset

  v8f acc[4][2][2] = {};

  constexpr int NS = D_IN / BK;     // 32 K-steps

  // prologue: stage slab 0 into buffer 0 (12 async ops per thread in flight)
  stage_async_tile(Xs[0], Hs[0], Ws[0], Xbf, Hbf, WtAll, tid, m0, n0, 0);

#pragma unroll 2
  for (int s = 0; s < NS; ++s) {
    const int cur = s & 1;
    const int nxt = cur ^ 1;
    if (s + 1 < NS) {
      // issue next slab first, then wait until only those 12 remain
      stage_async_tile(Xs[nxt], Hs[nxt], Ws[nxt], Xbf, Hbf, WtAll,
                       tid, m0, n0, (s + 1) * BK);
      wait_asynccnt_le12();
    } else {
      wait_asynccnt_0();
    }
    __syncthreads();   // all waves' current-slab data visible in LDS
    wmma_compute_step(Xs[cur], Hs[cur], Ws[cur],
                      wave_m, wave_n, lane15, khalfA, kbB, acc);
    __syncthreads();   // done reading cur before it is overwritten next iter
  }

  lstm_epilogue(acc, bf_, bi_, bo_, bc_, carousel, outH, outC,
                m0, n0, wave_m, wave_n, lane15, hi_half);
}

// ---------------------------------------------------------------------------
// Fallback: single-buffered, stages from f32 inputs converting on the fly
// ---------------------------------------------------------------------------
__global__ __launch_bounds__(256)
void lstm_fused_wmma_f32(const float* __restrict__ X, const float* __restrict__ H,
                         const int* __restrict__ carousel,
                         const float* __restrict__ Wfx, const float* __restrict__ Wfh, const float* __restrict__ bf_,
                         const float* __restrict__ Wix, const float* __restrict__ Wih, const float* __restrict__ bi_,
                         const float* __restrict__ Wox, const float* __restrict__ Woh, const float* __restrict__ bo_,
                         const float* __restrict__ Wcx, const float* __restrict__ Wch, const float* __restrict__ bc_,
                         float* __restrict__ outH, float* __restrict__ outC) {
  __shared__ bf16_t Xs[BM][LDK];
  __shared__ bf16_t Hs[BM][LDK];
  __shared__ bf16_t Ws[8][BN][LDK];

  const int tid  = threadIdx.x;
  const int wid  = tid >> 5;
  const int lane = tid & 31;
  const int lane15 = lane & 15;
  const int hi_half = lane >> 4;
  const int wave_m = wid & 3;
  const int wave_n = wid >> 2;
  const int m0 = blockIdx.y * BM;
  const int n0 = blockIdx.x * BN;

  const float* Wsrc[8] = {Wfx, Wfh, Wix, Wih, Wox, Woh, Wcx, Wch};

  const int khalfA = hi_half * 16;
  const int kbB    = hi_half * 16;

  v8f acc[4][2][2] = {};

  for (int k0 = 0; k0 < D_IN; k0 += BK) {
    __syncthreads();
    {
      const int row8 = tid >> 3;
      const int c4   = (tid & 7) * 4;
#pragma unroll
      for (int pass = 0; pass < 4; ++pass) {
        const int row = pass * 32 + row8;
        const float4 vx = *(const float4*)(X + (size_t)(m0 + row) * D_IN  + k0 + c4);
        const float4 vh = *(const float4*)(H + (size_t)(m0 + row) * D_OUT + k0 + c4);
        Xs[row][c4 + 0] = (bf16_t)vx.x; Xs[row][c4 + 1] = (bf16_t)vx.y;
        Xs[row][c4 + 2] = (bf16_t)vx.z; Xs[row][c4 + 3] = (bf16_t)vx.w;
        Hs[row][c4 + 0] = (bf16_t)vh.x; Hs[row][c4 + 1] = (bf16_t)vh.y;
        Hs[row][c4 + 2] = (bf16_t)vh.z; Hs[row][c4 + 3] = (bf16_t)vh.w;
      }
    }
#pragma unroll
    for (int w = 0; w < 8; ++w) {
      const float* Wp = Wsrc[w];
#pragma unroll
      for (int r = 0; r < 2; ++r) {
        const int idx = r * 256 + tid;
        const int kk  = idx >> 4;
        const int nn  = (idx & 15) * 4;
        const float4 v = *(const float4*)(Wp + (size_t)(k0 + kk) * D_OUT + n0 + nn);
        Ws[w][nn + 0][kk] = (bf16_t)v.x;
        Ws[w][nn + 1][kk] = (bf16_t)v.y;
        Ws[w][nn + 2][kk] = (bf16_t)v.z;
        Ws[w][nn + 3][kk] = (bf16_t)v.w;
      }
    }
    __syncthreads();
    wmma_compute_step(Xs, Hs, Ws, wave_m, wave_n, lane15, khalfA, kbB, acc);
  }

  lstm_epilogue(acc, bf_, bi_, bo_, bc_, carousel, outH, outC,
                m0, n0, wave_m, wave_n, lane15, hi_half);
}

extern "C" void kernel_launch(void* const* d_in, const int* in_sizes, int n_in,
                              void* d_out, int out_size, void* d_ws, size_t ws_size,
                              hipStream_t stream) {
  const float* X   = (const float*)d_in[0];
  const float* H   = (const float*)d_in[1];
  const int*   car = (const int*)d_in[2];
  const float* Wfx = (const float*)d_in[3];
  const float* Wfh = (const float*)d_in[4];
  const float* bf_ = (const float*)d_in[5];
  const float* Wix = (const float*)d_in[6];
  const float* Wih = (const float*)d_in[7];
  const float* bi_ = (const float*)d_in[8];
  const float* Wox = (const float*)d_in[9];
  const float* Woh = (const float*)d_in[10];
  const float* bo_ = (const float*)d_in[11];
  const float* Wcx = (const float*)d_in[12];
  const float* Wch = (const float*)d_in[13];
  const float* bc_ = (const float*)d_in[14];

  float* outH = (float*)d_out;                         // h: [16384, 1024]
  float* outC = outH + (size_t)B_ROWS * D_OUT;         // c: [16384, 1024]

  const size_t nX = (size_t)B_ROWS * D_IN;
  const size_t nH = (size_t)B_ROWS * D_OUT;
  const size_t nW = 8ull * D_IN * D_OUT;
  const size_t needed = (nX + nH + nW) * sizeof(bf16_t);

  dim3 grid(D_OUT / BN, B_ROWS / BM);                  // (16, 128)
  dim3 block(256);

  if (d_ws != nullptr && ws_size >= needed) {
    bf16_t* Xbf = (bf16_t*)d_ws;
    bf16_t* Hbf = Xbf + nX;
    bf16_t* Wt  = Hbf + nH;

    cvt_f32_to_bf16<<<(int)(nX / 8 / 256), 256, 0, stream>>>(X, Xbf, (int)(nX / 8));
    cvt_f32_to_bf16<<<(int)(nH / 8 / 256), 256, 0, stream>>>(H, Hbf, (int)(nH / 8));
    wt_transpose_bf16<<<dim3(D_OUT / 64, D_IN / 64, 8), 256, 0, stream>>>(
        Wfx, Wfh, Wix, Wih, Wox, Woh, Wcx, Wch, Wt);

    lstm_fused_wmma_async<<<grid, block, 0, stream>>>(
        car, bf_, bi_, bo_, bc_, Xbf, Hbf, Wt, outH, outC);
  } else {
    lstm_fused_wmma_f32<<<grid, block, 0, stream>>>(
        X, H, car,
        Wfx, Wfh, bf_, Wix, Wih, bi_, Wox, Woh, bo_, Wcx, Wch, bc_,
        outH, outC);
  }
}
